// LoFTR_74062416052668
// MI455X (gfx1250) — compile-verified, matrix operand order checked by reading, not run on hardware
//
#include <hip/hip_runtime.h>
#include <hip/hip_bf16.h>

// ---------------------------------------------------------------------------
// LoFTR forward for MI455X (gfx1250, wave32, WMMA).
// NHWC/token-major activations + fragment-prepacked weights so every WMMA
// operand is fed by 16-byte vector loads (global_load_b128 / ds_load_b128).
// Dual accumulators break WMMA->WMMA RAW hazards; conv uses a ballot'ed
// all-in-bounds fast path so border masking stays off the hot loop.
// ---------------------------------------------------------------------------

typedef __attribute__((ext_vector_type(16))) _Float16 v16h;
typedef __attribute__((ext_vector_type(8)))  _Float16 v8h;
typedef __attribute__((ext_vector_type(8)))  float    v8f;
typedef _Float16 h16;

#define LTOK 6400          // 80*80 tokens
#define DM   128           // d_model
#define DH   32            // head dim

// ---------------- WMMA helpers (wave32, 16x16x32 f16 -> f32) ----------------

__device__ __forceinline__ v8f wmma16(v16h a, v16h b, v8f c) {
  return __builtin_amdgcn_wmma_f32_16x16x32_f16(false, a, false, b,
                                                (short)0, c, false, false);
}

__device__ __forceinline__ v16h cat8(v8h lo, v8h hi) {
  return __builtin_shufflevector(lo, hi, 0, 1, 2, 3, 4, 5, 6, 7,
                                 8, 9, 10, 11, 12, 13, 14, 15);
}

// Fragment loader for row-contiguous sources (two b128 loads per lane).
// A-frag:  row index = M (lane&15), contiguous dim = K.   stride = row pitch.
// B-frag (K contiguous per column): row index = N, contiguous dim = K.
// Per ISA 7.12.2 interleave: lanes 0-15 kb=0, lanes 16-31 kb=8;
// elements {kb..kb+7} then {16+kb..16+kb+7}.
__device__ __forceinline__ v16h ldfrag_row(const h16* p, int stride) {
  int lane = threadIdx.x & 31;
  const h16* r = p + (lane & 15) * stride + ((lane >> 4) << 3);
  v8h lo = *(const v8h*)r;
  v8h hi = *(const v8h*)(r + 16);
  return cat8(lo, hi);
}

// B fragment from prepacked weights: 32 bytes contiguous per lane.
__device__ __forceinline__ v16h ldfragB_packed(const h16* Bp, int ntiles,
                                               int kt, int nt) {
  const h16* p = Bp + ((((kt * ntiles) + nt) << 5) + (threadIdx.x & 31)) * 16;
  v8h lo = *(const v8h*)p;
  v8h hi = *(const v8h*)(p + 8);
  return cat8(lo, hi);
}

// cross-lane reductions within a 16-lane half (rows of a C tile)
__device__ __forceinline__ float redmax16(float v) {
#pragma unroll
  for (int m = 1; m < 16; m <<= 1) v = fmaxf(v, __shfl_xor(v, m, 32));
  return v;
}
__device__ __forceinline__ float redsum16(float v) {
#pragma unroll
  for (int m = 1; m < 16; m <<= 1) v += __shfl_xor(v, m, 32);
  return v;
}

// ------------- weight prepack into WMMA B-fragment layout -------------------
// dense: src f32 [N][K] (nn.Linear weight); dst tile (kt,nt): lane-major 16h.

__global__ void k_packW(const float* __restrict__ src, h16* __restrict__ dst,
                        int N, int K) {
  int d = blockIdx.x * blockDim.x + threadIdx.x;
  if (d >= N * K) return;
  int e = d & 15, lane = (d >> 4) & 31, tile = d >> 9;
  int ntiles = N >> 4;
  int nt = tile % ntiles, kt = tile / ntiles;
  int kb = (lane >> 4) << 3;
  int k = (kt << 5) + ((e < 8) ? (kb + e) : (16 + kb + (e - 8)));
  int n = (nt << 4) + (lane & 15);
  dst[d] = (h16)src[n * K + k];
}

// conv: src f32 [Cout][Cin][3][3]; GEMM K order = tap*Cin + ci (tap-major,
// matching the NHWC im2col gather where channels are contiguous).
__global__ void k_packWconv(const float* __restrict__ src, h16* __restrict__ dst,
                            int Cout, int Cin) {
  int K = Cin * 9;
  int d = blockIdx.x * blockDim.x + threadIdx.x;
  if (d >= Cout * K) return;
  int e = d & 15, lane = (d >> 4) & 31, tile = d >> 9;
  int ntiles = Cout >> 4;
  int nt = tile % ntiles, kt = tile / ntiles;
  int kb = (lane >> 4) << 3;
  int k = (kt << 5) + ((e < 8) ? (kb + e) : (16 + kb + (e - 8)));
  int n = (nt << 4) + (lane & 15);
  int tap = k / Cin, ci = k - tap * Cin;
  dst[d] = (h16)src[(n * Cin + ci) * 9 + tap];
}

// ---------------- conv1: Cin=1, direct, fused BN+ReLU, NHWC f16 out ---------

__global__ void k_conv1(const float* __restrict__ img, const float* __restrict__ w,
                        const float* __restrict__ cb, const float* __restrict__ g,
                        const float* __restrict__ bb, h16* __restrict__ out) {
  int idx = blockIdx.x * blockDim.x + threadIdx.x;   // p*64 + c
  if (idx >= 320 * 320 * 64) return;
  int c = idx & 63, p = idx >> 6;
  int x = p % 320, y = p / 320;
  float acc = cb[c];
#pragma unroll
  for (int ky = 0; ky < 3; ++ky)
#pragma unroll
    for (int kx = 0; kx < 3; ++kx) {
      int yy = y + ky - 1, xx = x + kx - 1;
      if ((unsigned)yy < 320u && (unsigned)xx < 320u)
        acc += w[c * 9 + ky * 3 + kx] * img[yy * 320 + xx];
    }
  float s = g[c] * rsqrtf(1.0f + 1e-5f);
  out[idx] = (h16)fmaxf(acc * s + bb[c], 0.0f);
}

// ------- 3x3 conv as implicit-GEMM WMMA, fused BN+ReLU, NHWC f16 ------------
// in: [Hin*Win][Cin] f16; Bp: prepacked (K=Cin*9 tap-major); out: [Pout][Cout]

__global__ void k_conv(const h16* __restrict__ in, const h16* __restrict__ Bp,
                       const float* __restrict__ cb, const float* __restrict__ gg,
                       const float* __restrict__ bb, h16* __restrict__ out,
                       int Cin, int Cout, int Hin, int Win,
                       int Hout, int Wout, int stride) {
  int wt = blockIdx.x * (blockDim.x >> 5) + (threadIdx.x >> 5);
  int Pout = Hout * Wout;
  int nct = Cout >> 4;
  if (wt >= (Pout >> 4) * nct) return;
  int pt = wt / nct, ct = wt % nct;
  int lane = threadIdx.x & 31;
  int m = lane & 15, kb = (lane >> 4) << 3;
  int p = (pt << 4) + m;
  int oy = p / Wout, ox = p - oy * Wout;
  int iy0 = oy * stride - 1, ix0 = ox * stride - 1;
  v8f acc0 = {}, acc1 = {};        // dual accumulators: no WMMA->WMMA RAW
#pragma unroll
  for (int tap = 0; tap < 9; ++tap) {
    int iy = iy0 + tap / 3, ix = ix0 + tap % 3;
    bool inb = ((unsigned)iy < (unsigned)Hin) && ((unsigned)ix < (unsigned)Win);
    const h16* src = in + (iy * Win + ix) * Cin + kb;
    bool allin = ((unsigned)__ballot(inb)) == 0xffffffffu;  // wave-uniform
    if (allin) {
      // fast path: unguarded vector loads, EXEC stays all-1s
      for (int c0 = 0; c0 < Cin; c0 += 32) {
        v16h a = cat8(*(const v8h*)(src + c0), *(const v8h*)(src + c0 + 16));
        v16h b = ldfragB_packed(Bp, nct, (tap * Cin + c0) >> 5, ct);
        if (tap & 1) acc1 = wmma16(a, b, acc1);
        else         acc0 = wmma16(a, b, acc0);
      }
    } else {
      // border path: per-lane masked loads, zero padding
      for (int c0 = 0; c0 < Cin; c0 += 32) {
        v16h a = {};
        if (inb) a = cat8(*(const v8h*)(src + c0), *(const v8h*)(src + c0 + 16));
        v16h b = ldfragB_packed(Bp, nct, (tap * Cin + c0) >> 5, ct);
        if (tap & 1) acc1 = wmma16(a, b, acc1);
        else         acc0 = wmma16(a, b, acc0);
      }
    }
  }
  v8f c = acc0 + acc1;
  int n = (ct << 4) + (lane & 15);
  int mo = (pt << 4) + ((lane >> 4) << 3);
  float s = gg[n] * rsqrtf(1.0f + 1e-5f);
  float cbn = cb[n], bbn = bb[n];
#pragma unroll
  for (int r = 0; r < 8; ++r) {
    float v = fmaxf((c[r] + cbn) * s + bbn, 0.0f);
    out[(mo + r) * Cout + n] = (h16)v;
  }
}

// ------- lat3 1x1 conv (row-major GEMM) + bias + nearest-interp PE ----------
// a4: [6400][128] f16 token-major; Bp packed; pos: [128][32][32] f32

__global__ void k_lat3(const h16* __restrict__ a4, const h16* __restrict__ Bp,
                       const float* __restrict__ bias, const float* __restrict__ pos,
                       float* __restrict__ Ff, h16* __restrict__ Fh) {
  int wt = blockIdx.x * (blockDim.x >> 5) + (threadIdx.x >> 5);
  if (wt >= (LTOK >> 4) * (DM >> 4)) return;
  int m0 = (wt >> 3) << 4, nt = wt & 7, n0 = nt << 4;
  v8f c0 = {}, c1 = {};
#pragma unroll
  for (int k = 0; k < DM; k += 32) {
    v16h a = ldfrag_row(a4 + m0 * DM + k, DM);
    v16h b = ldfragB_packed(Bp, DM >> 4, k >> 5, nt);
    if ((k >> 5) & 1) c1 = wmma16(a, b, c1);
    else              c0 = wmma16(a, b, c0);
  }
  v8f c = c0 + c1;
  int lane = threadIdx.x & 31;
  int n = n0 + (lane & 15);
  int mo = m0 + ((lane >> 4) << 3);
  float bn = bias[n];
#pragma unroll
  for (int r = 0; r < 8; ++r) {
    int p = mo + r;
    int y = p / 80, x = p - y * 80;
    int ih = (y * 32) / 80, iw = (x * 32) / 80;
    float v = c[r] + bn + pos[n * 1024 + ih * 32 + iw];
    Ff[p * DM + n] = v;
    Fh[p * DM + n] = (h16)v;
  }
}

// ---------------- generic GEMM: C = act(A[MxK] * Bpacked + bias) ------------
// K must be a multiple of 64 (two K-steps per iteration, dual accumulators).

__global__ void k_gemm(const h16* __restrict__ A, const h16* __restrict__ Bp,
                       const float* __restrict__ bias, float* __restrict__ Cf,
                       h16* __restrict__ Ch, int M, int N, int K, int relu) {
  int wt = blockIdx.x * (blockDim.x >> 5) + (threadIdx.x >> 5);
  int nt = N >> 4;
  if (wt >= (M >> 4) * nt) return;
  int m0 = (wt / nt) << 4, ct = wt % nt;
  v8f c0 = {}, c1 = {};
  for (int k = 0; k < K; k += 64) {
    __builtin_prefetch(A + m0 * K + k + 64, 0, 1);   // global_prefetch_b8
    v16h a0 = ldfrag_row(A + m0 * K + k, K);
    v16h b0 = ldfragB_packed(Bp, nt, k >> 5, ct);
    v16h a1 = ldfrag_row(A + m0 * K + k + 32, K);
    v16h b1 = ldfragB_packed(Bp, nt, (k >> 5) + 1, ct);
    c0 = wmma16(a0, b0, c0);
    c1 = wmma16(a1, b1, c1);
  }
  v8f c = c0 + c1;
  int lane = threadIdx.x & 31;
  int n = (ct << 4) + (lane & 15);
  int mo = m0 + ((lane >> 4) << 3);
  float bs = bias ? bias[n] : 0.0f;
#pragma unroll
  for (int r = 0; r < 8; ++r) {
    float v = c[r] + bs;
    if (relu) v = fmaxf(v, 0.0f);
    if (Cf) Cf[(mo + r) * N + n] = v;
    if (Ch) Ch[(mo + r) * N + n] = (h16)v;
  }
}

// ---------------- V transpose: qkv[:,256+c] -> VT [128][L] ------------------

__global__ void k_trv(const h16* __restrict__ qkv, h16* __restrict__ vt) {
  int i = blockIdx.x * blockDim.x + threadIdx.x;
  if (i >= DM * LTOK) return;
  int p = i % LTOK, c = i / LTOK;
  vt[i] = qkv[p * 384 + 2 * DM + c];
}

// ---------------- flash attention: 4 heads, dh=32, L=6400 -------------------
// qkv: [L][384] f16 (q|k|v); vt: [128][L] f16; out: [L][128] f16
// block = 128 threads = 4 waves = 4 heads; one 16-query tile per block.

__global__ void k_attn(const h16* __restrict__ qkv, const h16* __restrict__ vt,
                       h16* __restrict__ out) {
  __shared__ h16 Pld[4 * 512];                 // per-wave 16x32 P tile
  int wave = threadIdx.x >> 5;
  int lane = threadIdx.x & 31;
  int q0 = blockIdx.x << 4;
  int qoff = wave * DH, koff = DM + wave * DH;
  v16h aQ = ldfrag_row(qkv + q0 * 384 + qoff, 384);
  v8f o0 = {}, o1 = {};
  float mrow[8], lrow[8];
#pragma unroll
  for (int r = 0; r < 8; ++r) { mrow[r] = -1e30f; lrow[r] = 0.0f; }
  h16* myP = Pld + wave * 512;
  int n = lane & 15, mo = (lane >> 4) << 3;
  const float sc = 0.1767766953f;              // 1/sqrt(32)

  for (int kv = 0; kv < LTOK; kv += 32) {
    // S = Q * K^T : B(k=d, n=key) -> K rows of qkv are K-contiguous
    v16h bK0 = ldfrag_row(qkv + kv * 384 + koff, 384);
    v16h bK1 = ldfrag_row(qkv + (kv + 16) * 384 + koff, 384);
    v8f s0 = {}, s1 = {};
    s0 = wmma16(aQ, bK0, s0);
    s1 = wmma16(aQ, bK1, s1);

    __syncthreads();                           // prior P reads done
#pragma unroll
    for (int r = 0; r < 8; ++r) {
      float v0 = s0[r] * sc, v1 = s1[r] * sc;
      float mx = redmax16(fmaxf(v0, v1));
      float mnew = fmaxf(mrow[r], mx);
      float alpha = __expf(mrow[r] - mnew);
      mrow[r] = mnew;
      float p0 = __expf(v0 - mnew), p1 = __expf(v1 - mnew);
      lrow[r] = lrow[r] * alpha + redsum16(p0 + p1);
      o0[r] *= alpha;
      o1[r] *= alpha;
      myP[(mo + r) * 32 + n]      = (h16)p0;
      myP[(mo + r) * 32 + 16 + n] = (h16)p1;
    }
    __syncthreads();                           // P visible for A-frag load

    // O += P * V : B(k=key, n=d); vt rows are key-contiguous
    v16h aP  = ldfrag_row(myP, 32);
    v16h bV0 = ldfrag_row(vt + qoff * LTOK + kv, LTOK);
    v16h bV1 = ldfrag_row(vt + (qoff + 16) * LTOK + kv, LTOK);
    o0 = wmma16(aP, bV0, o0);
    o1 = wmma16(aP, bV1, o1);
  }
#pragma unroll
  for (int r = 0; r < 8; ++r) {
    int p = q0 + mo + r;
    float inv = 1.0f / lrow[r];
    out[p * DM + qoff + n]      = (h16)(o0[r] * inv);
    out[p * DM + qoff + 16 + n] = (h16)(o1[r] * inv);
  }
}

// ---------------- residual + LayerNorm (one wave per 128-wide row) ----------

__global__ void k_resln(float* __restrict__ x, const float* __restrict__ hin,
                        const float* __restrict__ g, const float* __restrict__ b,
                        h16* __restrict__ xh, int M) {
  int row = blockIdx.x * (blockDim.x >> 5) + (threadIdx.x >> 5);
  if (row >= M) return;
  int lane = threadIdx.x & 31;
  float v[4];
  float s = 0.0f;
#pragma unroll
  for (int i = 0; i < 4; ++i) {
    int c = lane + 32 * i;
    v[i] = x[row * DM + c] + hin[row * DM + c];
    s += v[i];
  }
#pragma unroll
  for (int m = 1; m < 32; m <<= 1) s += __shfl_xor(s, m, 32);
  float mean = s * (1.0f / DM);
  float var = 0.0f;
#pragma unroll
  for (int i = 0; i < 4; ++i) { float d = v[i] - mean; var += d * d; }
#pragma unroll
  for (int m = 1; m < 32; m <<= 1) var += __shfl_xor(var, m, 32);
  float inv = rsqrtf(var * (1.0f / DM) + 1e-5f);
#pragma unroll
  for (int i = 0; i < 4; ++i) {
    int c = lane + 32 * i;
    float o = (v[i] - mean) * inv * g[c] + b[c];
    x[row * DM + c] = o;
    xh[row * DM + c] = (h16)o;
  }
}

// ---------------- correlation + fused softmax over j (len 80) ---------------
// wave tile: 16 hw-rows x 80 cols (one full softmax group), K = 128.

__global__ void k_corr(const h16* __restrict__ F0, const h16* __restrict__ F1,
                       float* __restrict__ out) {
  int wt = blockIdx.x * (blockDim.x >> 5) + (threadIdx.x >> 5);
  if (wt >= 400 * 80) return;
  int m0 = (wt / 80) << 4;
  int n0 = (wt % 80) * 80;
  v16h a[4];
#pragma unroll
  for (int kk = 0; kk < 4; ++kk) a[kk] = ldfrag_row(F0 + m0 * DM + kk * 32, DM);
  v8f c[5];
#pragma unroll
  for (int j = 0; j < 5; ++j) {
    v8f e0 = {}, e1 = {};
#pragma unroll
    for (int kk = 0; kk < 4; ++kk) {
      // B(k, n) = F1[n0+j*16+n][k] : rows of F1 are K-contiguous
      v16h b = ldfrag_row(F1 + (n0 + j * 16) * DM + kk * 32, DM);
      if (kk & 1) e1 = wmma16(a[kk], b, e1);
      else        e0 = wmma16(a[kk], b, e0);
    }
    c[j] = e0 + e1;
  }
  int lane = threadIdx.x & 31;
  int n = lane & 15, mo = (lane >> 4) << 3;
#pragma unroll
  for (int r = 0; r < 8; ++r) {
    float mx = -1e30f;
#pragma unroll
    for (int j = 0; j < 5; ++j) mx = fmaxf(mx, c[j][r]);
    mx = redmax16(mx);
    float e[5], s = 0.0f;
#pragma unroll
    for (int j = 0; j < 5; ++j) { e[j] = __expf((c[j][r] - mx) * 10.0f); s += e[j]; }
    s = redsum16(s);
    float inv = 1.0f / s;
    long long base = (long long)(m0 + mo + r) * LTOK + n0;
#pragma unroll
    for (int j = 0; j < 5; ++j) out[base + j * 16 + n] = e[j] * inv;
  }
}

// ---------------- [L][C] f32 -> [C][L] f32 (final f0/f1 outputs) ------------

__global__ void k_tr(const float* __restrict__ F, float* __restrict__ out) {
  int i = blockIdx.x * blockDim.x + threadIdx.x;
  if (i >= DM * LTOK) return;
  int p = i % LTOK;                // out is [C][L]
  int c = i / LTOK;
  out[i] = F[p * DM + c];
}

// ===========================================================================

extern "C" void kernel_launch(void* const* d_in, const int* in_sizes, int n_in,
                              void* d_out, int out_size, void* d_ws, size_t ws_size,
                              hipStream_t stream) {
  (void)in_sizes; (void)n_in; (void)out_size; (void)ws_size;

  // input index map (setup_inputs dict order, depth-first)
  const float* image[2] = { (const float*)d_in[0], (const float*)d_in[1] };
  const float* conv1_w = (const float*)d_in[2];
  const float* conv1_b = (const float*)d_in[3];
  const float* bn1_g   = (const float*)d_in[4];
  const float* bn1_b   = (const float*)d_in[5];
  const float* l1_w = (const float*)d_in[6];  const float* l1_b = (const float*)d_in[7];
  const float* l1_g = (const float*)d_in[8];  const float* l1_bb = (const float*)d_in[9];
  const float* l2_w = (const float*)d_in[10]; const float* l2_b = (const float*)d_in[11];
  const float* l2_g = (const float*)d_in[12]; const float* l2_bb = (const float*)d_in[13];
  const float* l3_w = (const float*)d_in[14]; const float* l3_b = (const float*)d_in[15];
  const float* l3_g = (const float*)d_in[16]; const float* l3_bb = (const float*)d_in[17];
  const float* lat3_w = (const float*)d_in[18];
  const float* lat3_b = (const float*)d_in[19];
  const float* pos    = (const float*)d_in[20];
  const int LBASE = 21;   // + layer*12: in_w,in_b,out_w,out_b,w1,b1,w2,b2,n1_g,n1_b,n2_g,n2_b

  // deterministic bump allocator over d_ws
  char* wsb = (char*)d_ws;
  size_t off = 0;
  auto alloc = [&](size_t bytes) -> void* {
    void* p = wsb + off;
    off = (off + bytes + 255) & ~(size_t)255;
    return p;
  };

  // conv scratch, NHWC f16 (reused across the two images)
  h16* a1 = (h16*)alloc((size_t)102400 * 64 * 2);
  h16* a2 = (h16*)alloc((size_t)102400 * 64 * 2);
  h16* a3 = (h16*)alloc((size_t)25600 * 128 * 2);
  h16* a4 = (h16*)alloc((size_t)6400 * 128 * 2);
  // per-image token features
  float* Ff[2]; h16* Fh[2];
  for (int i = 0; i < 2; ++i) {
    Ff[i] = (float*)alloc((size_t)LTOK * DM * 4);
    Fh[i] = (h16*)alloc((size_t)LTOK * DM * 2);
  }
  // transformer scratch (reused)
  h16*  qkvh = (h16*)alloc((size_t)LTOK * 384 * 2);
  h16*  vth  = (h16*)alloc((size_t)DM * LTOK * 2);
  h16*  atth = (h16*)alloc((size_t)LTOK * DM * 2);
  float* projf = (float*)alloc((size_t)LTOK * DM * 4);
  h16*  ffnh = (h16*)alloc((size_t)LTOK * DM * 2);
  float* ffnf = (float*)alloc((size_t)LTOK * DM * 4);
  // prepacked f16 weights (WMMA B-fragment layout)
  h16* l1P   = (h16*)alloc((size_t)576 * 64 * 2);
  h16* l2P   = (h16*)alloc((size_t)576 * 128 * 2);
  h16* l3P   = (h16*)alloc((size_t)1152 * 128 * 2);
  h16* lat3P = (h16*)alloc((size_t)128 * 128 * 2);
  h16 *inP[4], *outP[4], *w1P[4], *w2P[4];
  for (int l = 0; l < 4; ++l) {
    inP[l]  = (h16*)alloc((size_t)128 * 384 * 2);
    outP[l] = (h16*)alloc((size_t)128 * 128 * 2);
    w1P[l]  = (h16*)alloc((size_t)128 * 128 * 2);
    w2P[l]  = (h16*)alloc((size_t)128 * 128 * 2);
  }

  auto cdiv = [](int a, int b) { return (a + b - 1) / b; };

  // --- weight prepacks ---
  k_packWconv<<<cdiv(64 * 576, 256), 256, 0, stream>>>(l1_w, l1P, 64, 64);
  k_packWconv<<<cdiv(128 * 576, 256), 256, 0, stream>>>(l2_w, l2P, 128, 64);
  k_packWconv<<<cdiv(128 * 1152, 256), 256, 0, stream>>>(l3_w, l3P, 128, 128);
  k_packW<<<cdiv(128 * 128, 256), 256, 0, stream>>>(lat3_w, lat3P, 128, 128);
  for (int l = 0; l < 4; ++l) {
    const float* in_w  = (const float*)d_in[LBASE + l * 12 + 0];
    const float* out_w = (const float*)d_in[LBASE + l * 12 + 2];
    const float* w1    = (const float*)d_in[LBASE + l * 12 + 4];
    const float* w2    = (const float*)d_in[LBASE + l * 12 + 6];
    k_packW<<<cdiv(384 * 128, 256), 256, 0, stream>>>(in_w, inP[l], 384, 128);
    k_packW<<<cdiv(128 * 128, 256), 256, 0, stream>>>(out_w, outP[l], 128, 128);
    k_packW<<<cdiv(128 * 128, 256), 256, 0, stream>>>(w1, w1P[l], 128, 128);
    k_packW<<<cdiv(128 * 128, 256), 256, 0, stream>>>(w2, w2P[l], 128, 128);
  }

  // --- backbone per image (all activations NHWC) ---
  for (int img = 0; img < 2; ++img) {
    k_conv1<<<cdiv(102400 * 64, 256), 256, 0, stream>>>(
        image[img], conv1_w, conv1_b, bn1_g, bn1_b, a1);
    // l1: 64->64, 320x320, stride 1 : waves = 6400*4
    k_conv<<<cdiv(6400 * 4, 8), 256, 0, stream>>>(
        a1, l1P, l1_b, l1_g, l1_bb, a2, 64, 64, 320, 320, 320, 320, 1);
    // l2: 64->128, ->160x160, stride 2 : waves = 1600*8
    k_conv<<<cdiv(1600 * 8, 8), 256, 0, stream>>>(
        a2, l2P, l2_b, l2_g, l2_bb, a3, 64, 128, 320, 320, 160, 160, 2);
    // l3: 128->128, ->80x80, stride 2 : waves = 400*8
    k_conv<<<cdiv(400 * 8, 8), 256, 0, stream>>>(
        a3, l3P, l3_b, l3_g, l3_bb, a4, 128, 128, 160, 160, 80, 80, 2);
    // lat3 1x1 + positional encoding -> token-major features
    k_lat3<<<cdiv(400 * 8, 8), 256, 0, stream>>>(
        a4, lat3P, lat3_b, pos, Ff[img], Fh[img]);
  }

  // --- transformer layers ---
  for (int img = 0; img < 2; ++img) {
    for (int l = 0; l < 4; ++l) {
      const float* in_b  = (const float*)d_in[LBASE + l * 12 + 1];
      const float* out_b = (const float*)d_in[LBASE + l * 12 + 3];
      const float* b1    = (const float*)d_in[LBASE + l * 12 + 5];
      const float* b2    = (const float*)d_in[LBASE + l * 12 + 7];
      const float* n1g   = (const float*)d_in[LBASE + l * 12 + 8];
      const float* n1b   = (const float*)d_in[LBASE + l * 12 + 9];
      const float* n2g   = (const float*)d_in[LBASE + l * 12 + 10];
      const float* n2b   = (const float*)d_in[LBASE + l * 12 + 11];

      // QKV projection: [6400,128] x [128,384]
      k_gemm<<<cdiv(400 * 24, 8), 256, 0, stream>>>(
          Fh[img], inP[l], in_b, (float*)nullptr, qkvh, LTOK, 384, DM, 0);
      // V transpose for contiguous PV B-fragments
      k_trv<<<cdiv(DM * LTOK, 256), 256, 0, stream>>>(qkvh, vth);
      // flash attention
      k_attn<<<400, 128, 0, stream>>>(qkvh, vth, atth);
      // output projection
      k_gemm<<<cdiv(400 * 8, 8), 256, 0, stream>>>(
          atth, outP[l], out_b, projf, (h16*)nullptr, LTOK, DM, DM, 0);
      // x = LN(x + proj)
      k_resln<<<cdiv(LTOK, 8), 256, 0, stream>>>(Ff[img], projf, n1g, n1b, Fh[img], LTOK);
      // FFN
      k_gemm<<<cdiv(400 * 8, 8), 256, 0, stream>>>(
          Fh[img], w1P[l], b1, (float*)nullptr, ffnh, LTOK, DM, DM, 1);
      k_gemm<<<cdiv(400 * 8, 8), 256, 0, stream>>>(
          ffnh, w2P[l], b2, ffnf, (h16*)nullptr, LTOK, DM, DM, 0);
      // x = LN(x + ffn)
      k_resln<<<cdiv(LTOK, 8), 256, 0, stream>>>(Ff[img], ffnf, n2g, n2b, Fh[img], LTOK);
    }
  }

  // --- correlation + fused softmax -> scores (d_out[0 .. 6400*6400)) ---
  float* scores = (float*)d_out;
  k_corr<<<cdiv(400 * 80, 8), 256, 0, stream>>>(Fh[0], Fh[1], scores);

  // --- final f0/f1 as [1,128,80,80] f32 appended after scores ---
  float* f0_out = scores + (size_t)LTOK * LTOK;
  float* f1_out = f0_out + (size_t)DM * LTOK;
  k_tr<<<cdiv(DM * LTOK, 256), 256, 0, stream>>>(Ff[0], f0_out);
  k_tr<<<cdiv(DM * LTOK, 256), 256, 0, stream>>>(Ff[1], f1_out);
}
